// SimplifiedSACB_75900662055126
// MI455X (gfx1250) — compile-verified, matrix operand order checked by reading, not run on hardware
//
#include <hip/hip_runtime.h>
#include <hip/hip_bf16.h>

typedef __attribute__((ext_vector_type(16))) _Float16 v16h;
typedef __attribute__((ext_vector_type(8)))  _Float16 v8h;
typedef __attribute__((ext_vector_type(8)))  float    v8f;

#define DD 96
#define HH 96
#define WW 96
#define NC 32          // channels (in == out == 32)
#define LWP 98         // LDS row width (96 + 2 halo)
#define LCP 32         // LDS channel stride (halfs)
#define RCHUNK 384     // reduction chunks per batch image

// ---------------------------------------------------------------------------
// Prep: zero stats, run the tiny K=4 MLPs (mod + bias), pack conv weights into
// the per-lane WMMA A-fragment layout ([27 off][2 tiles][32 lanes][16 halfs]).
// ---------------------------------------------------------------------------
__global__ __launch_bounds__(256)
void prep_kernel(const float* __restrict__ cc,
                 const float* __restrict__ proj_w,
                 const float* __restrict__ base_w,
                 const float* __restrict__ w1, const float* __restrict__ b1,
                 const float* __restrict__ w2, const float* __restrict__ b2,
                 const float* __restrict__ w3, const float* __restrict__ b3,
                 const float* __restrict__ g1w, const float* __restrict__ g1b,
                 const float* __restrict__ g2w, const float* __restrict__ g2b,
                 _Float16* __restrict__ wpackP, _Float16* __restrict__ wpackM,
                 float* __restrict__ modp, float* __restrict__ biasp,
                 float* __restrict__ stats1, float* __restrict__ stats2)
{
    const int tid = threadIdx.x;
    if (tid < 128) stats1[tid] = 0.f; else stats2[tid - 128] = 0.f;

    __shared__ float sh1[4][128];
    __shared__ float sh2[4][128];
    __shared__ float sg[4][64];

    for (int k = 0; k < 4; ++k) {
        if (tid < 128) {
            float s = b1[tid];
            for (int c = 0; c < 32; ++c) s += w1[tid * 32 + c] * cc[k * 32 + c];
            sh1[k][tid] = fmaxf(s, 0.f);
        }
        __syncthreads();
        if (tid < 128) {
            float s = b2[tid];
            for (int j = 0; j < 128; ++j) s += w2[tid * 128 + j] * sh1[k][j];
            sh2[k][tid] = fmaxf(s, 0.f);
        }
        if (tid >= 128 && tid < 192) {
            int i = tid - 128;
            float s = g1b[i];
            for (int c = 0; c < 32; ++c) s += g1w[i * 32 + c] * cc[k * 32 + c];
            sg[k][i] = fmaxf(s, 0.f);
        }
        __syncthreads();
        if (tid < 27) {
            float s = b3[tid];
            for (int j = 0; j < 128; ++j) s += w3[tid * 128 + j] * sh2[k][j];
            modp[k * 27 + tid] = 1.f / (1.f + __expf(-s));
        }
        if (tid >= 32 && tid < 64) {
            int o = tid - 32;
            float s = g2b[o];
            for (int j = 0; j < 64; ++j) s += g2w[o * 64 + j] * sg[k][j];
            biasp[k * 32 + o] = s;
        }
        __syncthreads();
    }

    // A-fragment packing (16-bit A 16x32 layout: lane m=lane%16, half h=lane/16,
    // halfs j<8 -> K = h*8+j, halfs j>=8 -> K = 16 + h*8 + (j-8))
    for (int idx = tid; idx < 27 * 2 * 32 * 16; idx += 256) {
        int j    = idx & 15;
        int lane = (idx >> 4) & 31;
        int tile = (idx >> 9) & 1;
        int off  = idx >> 10;
        int m  = lane & 15;
        int hs = lane >> 4;
        int k  = (j < 8) ? (hs * 8 + j) : (16 + hs * 8 + (j - 8));
        int o  = tile * 16 + m;
        wpackP[idx] = (_Float16)proj_w[(o * 32 + k) * 27 + off];
        wpackM[idx] = (_Float16)base_w[(o * 32 + k) * 27 + off];
    }
}

// ---------------------------------------------------------------------------
// x [b][c][d][h][w] fp32  ->  xh [b][d][h][w][c] f16 (voxel-major)
// ---------------------------------------------------------------------------
__global__ __launch_bounds__(256)
void convert_x_kernel(const float* __restrict__ x, _Float16* __restrict__ xh)
{
    const long DHW = (long)DD * HH * WW;
    long vox = (long)blockIdx.x * 256 + threadIdx.x;
    int  b   = (int)(vox / DHW);
    long r   = vox - (long)b * DHW;
#pragma unroll
    for (int c = 0; c < 32; ++c)
        xh[vox * 32 + c] = (_Float16)x[((long)b * 32 + c) * DHW + r];
}

// ---------------------------------------------------------------------------
// Implicit-GEMM 3x3x3 conv, C=32 -> O=32, via v_wmma_f32_16x16x32_f16.
// Block = one (b,d,h) row (96 voxels), 6 waves, wave = 16-voxel x 32-O tile.
// MAIN: scales the B fragment per lane by eff_mod(p,off)=sum_k cw_k*mod_k and
// seeds the accumulator with the cw-blended bias (collapses the 4-expert sum).
// ---------------------------------------------------------------------------
template<bool MAIN>
__global__ __launch_bounds__(192)
void conv3x3_wmma(const _Float16* __restrict__ in,     // [b][d][h][w][32] f16
                  const _Float16* __restrict__ wpack,  // packed A fragments
                  const float*    __restrict__ cwp,    // [b][d][h][w][4]
                  const float*    __restrict__ modp,   // [4][27]
                  const float*    __restrict__ biasp,  // [4][32]
                  float*          __restrict__ out)    // [b][d][h][w][32]
{
    __shared__ _Float16 smem[9 * LWP * LCP];           // 56448 B

    const int blk = blockIdx.x;
    const int h  = blk % HH;
    const int t2 = blk / HH;
    const int d  = t2 % DD;
    const int b  = t2 / DD;
    const int tid = threadIdx.x;

    // Fill halo tile: 9 (dz,dy) rows x 98 w x 32 c, in 16-byte chunks.
    for (int i = tid; i < 9 * LWP * 4; i += 192) {
        int c8  = i & 3;
        int r   = i >> 2;
        int lw  = r % LWP;
        int row = r / LWP;
        int dz = row / 3 - 1;
        int dy = row % 3 - 1;
        int gd = d + dz, gh = h + dy, gw = lw - 1;
        uint4 v = {0u, 0u, 0u, 0u};
        if (gd >= 0 && gd < DD && gh >= 0 && gh < HH && gw >= 0 && gw < WW) {
            long idx = ((((long)b * DD + gd) * HH + gh) * WW + gw) * NC + c8 * 8;
            v = *(const uint4*)(in + idx);
        }
        *(uint4*)(&smem[(row * LWP + lw) * LCP + c8 * 8]) = v;
    }
    __syncthreads();

    const int wave = tid >> 5;
    const int lane = tid & 31;
    const int n    = lane & 15;          // N position (voxel) in tile
    const int hsel = lane >> 4;          // lane-half selector
    const int w0   = wave * 16;
    const int wv   = w0 + n;
    const long vox = (((long)b * DD + d) * HH + h) * WW + wv;

    float e[27];
    v8f acc0 = {};
    v8f acc1 = {};
    if (MAIN) {
        const float c0 = cwp[vox * 4 + 0];
        const float c1 = cwp[vox * 4 + 1];
        const float c2 = cwp[vox * 4 + 2];
        const float c3 = cwp[vox * 4 + 3];
#pragma unroll
        for (int o = 0; o < 27; ++o)
            e[o] = c0 * modp[o] + c1 * modp[27 + o] + c2 * modp[54 + o] + c3 * modp[81 + o];
        const int ob0 = hsel * 8;
        const int ob1 = 16 + hsel * 8;
#pragma unroll
        for (int j = 0; j < 8; ++j) {
            acc0[j] = c0 * biasp[ob0 + j] + c1 * biasp[32 + ob0 + j]
                    + c2 * biasp[64 + ob0 + j] + c3 * biasp[96 + ob0 + j];
            acc1[j] = c0 * biasp[ob1 + j] + c1 * biasp[32 + ob1 + j]
                    + c2 * biasp[64 + ob1 + j] + c3 * biasp[96 + ob1 + j];
        }
    }

#pragma unroll
    for (int row = 0; row < 9; ++row) {
#pragma unroll
        for (int dxi = 0; dxi < 3; ++dxi) {
            const int off = row * 3 + dxi;
            // B fragment: K=c (32) x N=16 voxels; lane -> (n, K-half hsel)
            const _Float16* bp = &smem[(row * LWP + (w0 + n + dxi)) * LCP];
            v8h lo = *(const v8h*)(bp + hsel * 8);
            v8h hi = *(const v8h*)(bp + 16 + hsel * 8);
            v16h bf = __builtin_shufflevector(lo, hi,
                        0,1,2,3,4,5,6,7,8,9,10,11,12,13,14,15);
            if (MAIN) {
                _Float16 s = (_Float16)e[off];
                bf = bf * s;                 // per-lane == per-voxel eff_mod scale
            }
            v16h a0 = *(const v16h*)(wpack + ((long)(off * 2 + 0) * 32 + lane) * 16);
            v16h a1 = *(const v16h*)(wpack + ((long)(off * 2 + 1) * 32 + lane) * 16);
            acc0 = __builtin_amdgcn_wmma_f32_16x16x32_f16(false, a0, false, bf,
                                                          (short)0, acc0, false, false);
            acc1 = __builtin_amdgcn_wmma_f32_16x16x32_f16(false, a1, false, bf,
                                                          (short)0, acc1, false, false);
        }
    }

    // C/D layout: lane half -> M offset 0/8; VGPR j -> consecutive O.
    *(v8f*)(out + vox * 32 + hsel * 8)      = acc0;
    *(v8f*)(out + vox * 32 + 16 + hsel * 8) = acc1;
}

// ---------------------------------------------------------------------------
// Per-(b,c) sum/sumsq over voxel-major [b][d][h][w][32] fp32 tensor.
// ---------------------------------------------------------------------------
__global__ __launch_bounds__(256)
void stat_reduce(const float* __restrict__ y, float* __restrict__ stats)
{
    __shared__ float ls[256], lq[256];
    const int tid = threadIdx.x;
    const int b     = blockIdx.x / RCHUNK;
    const int chunk = blockIdx.x % RCHUNK;
    const long perB = (long)DD * HH * WW * 32;
    const long span = perB / RCHUNK;                 // 73728 (mult. of 256 & 32)
    const long base = (long)b * perB + (long)chunk * span;
    float s = 0.f, q = 0.f;
    for (long i = tid; i < span; i += 256) {
        float v = y[base + i];
        s += v; q += v * v;
    }
    ls[tid] = s; lq[tid] = q;
    __syncthreads();
    if (tid < 32) {                                  // channel c == tid
        for (int g = 1; g < 8; ++g) { s += ls[tid + 32 * g]; q += lq[tid + 32 * g]; }
        atomicAdd(&stats[(b * 32 + tid) * 2 + 0], s);
        atomicAdd(&stats[(b * 32 + tid) * 2 + 1], q);
    }
}

__global__ void stat_final(const float* __restrict__ stats, float* __restrict__ mr)
{
    const int i = threadIdx.x;                       // 64 = B*C
    const float N = (float)((long)DD * HH * WW);
    float m   = stats[i * 2 + 0] / N;
    float var = stats[i * 2 + 1] / N - m * m;
    mr[i * 2 + 0] = m;
    mr[i * 2 + 1] = rsqrtf(var + 1e-5f);
}

// ---------------------------------------------------------------------------
// InstanceNorm + LeakyReLU -> xp (f16) fused with cluster softmax -> cw.
// ---------------------------------------------------------------------------
__global__ __launch_bounds__(256)
void norm_cluster(const float* __restrict__ y, const float* __restrict__ mr,
                  const float* __restrict__ cc, _Float16* __restrict__ xp,
                  float* __restrict__ cw)
{
    const long DHW = (long)DD * HH * WW;
    long vox = (long)blockIdx.x * 256 + threadIdx.x;
    int  b   = (int)(vox / DHW);
    float s0 = 0.f, s1 = 0.f, s2 = 0.f, s3 = 0.f;
#pragma unroll
    for (int c = 0; c < 32; ++c) {
        float v = y[vox * 32 + c];
        v = (v - mr[(b * 32 + c) * 2]) * mr[(b * 32 + c) * 2 + 1];
        v = v >= 0.f ? v : 0.1f * v;
        xp[vox * 32 + c] = (_Float16)v;
        s0 += v * cc[c];      s1 += v * cc[32 + c];
        s2 += v * cc[64 + c]; s3 += v * cc[96 + c];
    }
    const float sc = 0.17677669529663687f;           // 1/sqrt(32)
    s0 *= sc; s1 *= sc; s2 *= sc; s3 *= sc;
    float mx = fmaxf(fmaxf(s0, s1), fmaxf(s2, s3));
    float e0 = __expf(s0 - mx), e1 = __expf(s1 - mx);
    float e2 = __expf(s2 - mx), e3 = __expf(s3 - mx);
    float inv = 1.f / (e0 + e1 + e2 + e3);
    cw[vox * 4 + 0] = e0 * inv; cw[vox * 4 + 1] = e1 * inv;
    cw[vox * 4 + 2] = e2 * inv; cw[vox * 4 + 3] = e3 * inv;
}

// ---------------------------------------------------------------------------
// Final InstanceNorm + LeakyReLU + residual; transpose back to [b][o][d][h][w].
// ---------------------------------------------------------------------------
__global__ __launch_bounds__(256)
void final_out(const float* __restrict__ opre, const float* __restrict__ mr,
               const float* __restrict__ x, float* __restrict__ outp)
{
    const long DHW = (long)DD * HH * WW;
    long vox = (long)blockIdx.x * 256 + threadIdx.x;
    int  b   = (int)(vox / DHW);
    long r   = vox - (long)b * DHW;
#pragma unroll
    for (int o = 0; o < 32; ++o) {
        float v = opre[vox * 32 + o];
        v = (v - mr[(b * 32 + o) * 2]) * mr[(b * 32 + o) * 2 + 1];
        v = v >= 0.f ? v : 0.1f * v;
        long gi = ((long)b * 32 + o) * DHW + r;
        outp[gi] = v + x[gi];
    }
}

// ---------------------------------------------------------------------------
extern "C" void kernel_launch(void* const* d_in, const int* in_sizes, int n_in,
                              void* d_out, int out_size, void* d_ws, size_t ws_size,
                              hipStream_t stream)
{
    (void)in_sizes; (void)n_in; (void)out_size; (void)ws_size;
    const float* x     = (const float*)d_in[0];
    const float* cc    = (const float*)d_in[1];
    const float* pw    = (const float*)d_in[2];
    const float* bw    = (const float*)d_in[3];
    const float* wm_w1 = (const float*)d_in[4];
    const float* wm_b1 = (const float*)d_in[5];
    const float* wm_w2 = (const float*)d_in[6];
    const float* wm_b2 = (const float*)d_in[7];
    const float* wm_w3 = (const float*)d_in[8];
    const float* wm_b3 = (const float*)d_in[9];
    const float* bg_w1 = (const float*)d_in[10];
    const float* bg_b1 = (const float*)d_in[11];
    const float* bg_w2 = (const float*)d_in[12];
    const float* bg_b2 = (const float*)d_in[13];

    char* ws = (char*)d_ws;
    _Float16* xh    = (_Float16*)(ws + 0);            // 113246208 B
    _Float16* xpH   = (_Float16*)(ws + 113246208);    // 113246208 B
    float*    ybuf  = (float*)(ws + 226492416);       // 226492416 B (y, then opre)
    float*    cwbuf = (float*)(ws + 452984832);       // 28311552 B
    _Float16* wpP   = (_Float16*)(ws + 481296384);    // 55296 B
    _Float16* wpM   = (_Float16*)(ws + 481351680);    // 55296 B
    float*    modb  = (float*)(ws + 481406976);       // 432 B
    float*    biasb = (float*)(ws + 481407488);       // 512 B
    float*    st1   = (float*)(ws + 481408000);       // 512 B
    float*    mr1   = (float*)(ws + 481408512);       // 512 B
    float*    st2   = (float*)(ws + 481409024);       // 512 B
    float*    mr2   = (float*)(ws + 481409536);       // 512 B

    const int nvoxblk = 2 * DD * HH * WW / 256;       // 6912
    const int convblk = 2 * DD * HH;                  // 18432

    prep_kernel<<<1, 256, 0, stream>>>(cc, pw, bw, wm_w1, wm_b1, wm_w2, wm_b2,
                                       wm_w3, wm_b3, bg_w1, bg_b1, bg_w2, bg_b2,
                                       wpP, wpM, modb, biasb, st1, st2);
    convert_x_kernel<<<nvoxblk, 256, 0, stream>>>(x, xh);
    conv3x3_wmma<false><<<convblk, 192, 0, stream>>>(xh, wpP, nullptr, nullptr,
                                                     nullptr, ybuf);
    stat_reduce<<<2 * RCHUNK, 256, 0, stream>>>(ybuf, st1);
    stat_final<<<1, 64, 0, stream>>>(st1, mr1);
    norm_cluster<<<nvoxblk, 256, 0, stream>>>(ybuf, mr1, cc, xpH, cwbuf);
    conv3x3_wmma<true><<<convblk, 192, 0, stream>>>(xpH, wpM, cwbuf, modb,
                                                    biasb, ybuf);
    stat_reduce<<<2 * RCHUNK, 256, 0, stream>>>(ybuf, st2);
    stat_final<<<1, 64, 0, stream>>>(st2, mr2);
    final_out<<<nvoxblk, 256, 0, stream>>>(ybuf, mr2, x, (float*)d_out);
}
